// PhiAttention_18648747999331
// MI455X (gfx1250) — matrix operand order repaired and checked by reading, not compiled, not run on hardware
//
#include <hip/hip_runtime.h>

// ---------------- problem constants ----------------
#define BB      2
#define SS      2048
#define HID     2560
#define NHEADS  32
#define HDIM    80
#define HDP     96          // head dim padded to multiple of 32 for WMMA K
#define QKV_N   (3*HID)     // 7680
#define MROWS   (BB*SS)     // 4096

typedef __bf16 bf16x16 __attribute__((ext_vector_type(16)));
typedef float  v8f     __attribute__((ext_vector_type(8)));

union Frag { bf16x16 v; unsigned int u[8]; };
union U4u16 { uint4 q; unsigned short s[8]; };

__device__ __forceinline__ v8f zero8() {
    v8f z;
    #pragma unroll
    for (int i = 0; i < 8; i++) z[i] = 0.f;
    return z;
}

__device__ __forceinline__ unsigned short f2bf(float f) {
    unsigned int u = __builtin_bit_cast(unsigned int, f);
    unsigned int r = u + 0x7FFFu + ((u >> 16) & 1u);   // round-to-nearest-even
    return (unsigned short)(r >> 16);
}
__device__ __forceinline__ float bf2f(unsigned short b) {
    unsigned int u = ((unsigned int)b) << 16;
    return __builtin_bit_cast(float, u);
}

// ---------------- fp32 -> bf16 convert ----------------
__global__ __launch_bounds__(256) void cvt_f32_bf16(const float* __restrict__ src,
                                                    unsigned short* __restrict__ dst,
                                                    size_t n) {
    size_t i = (size_t)blockIdx.x * blockDim.x + threadIdx.x;
    if (i < n) dst[i] = f2bf(src[i]);
}

// ---------------- bf16 GEMM: C[M,N] = A[M,K] * B[K,N] + bias ----------------
// 128x128 block tile, K step 32, 256 threads (8 waves), each wave 64x32 (8 WMMA tiles)
#define BM 128
#define BN 128
#define BK 32
#define LSTR 40   // LDS row stride (elems): 80B rows -> 16B aligned vector ops

__device__ __forceinline__ void store_out(float* C, size_t i, float v)          { C[i] = v; }
__device__ __forceinline__ void store_out(unsigned short* C, size_t i, float v) { C[i] = f2bf(v); }

template <typename OutT>
__global__ __launch_bounds__(256)
void gemm_bf16(const unsigned short* __restrict__ A,
               const unsigned short* __restrict__ B,
               const float* __restrict__ bias,
               OutT* __restrict__ C,
               int M, int N, int K) {
    __shared__ unsigned short As[BM * LSTR];
    __shared__ unsigned short Bs[BN * LSTR];   // stored transposed: Bs[n][k]

    const int t    = threadIdx.x;
    const int lane = t & 31;
    const int wid  = t >> 5;
    const int wr   = wid >> 2;          // 0..1  -> 64-row strip
    const int wc   = wid & 3;           // 0..3  -> 32-col strip
    const int m0   = blockIdx.y * BM;
    const int n0   = blockIdx.x * BN;

    const int a_row  = t >> 1;          // 0..127
    const int a_ks   = (t & 1) * 16;    // 0 or 16
    const int b_krow = t >> 3;          // 0..31
    const int b_ns   = (t & 7) * 16;    // 0..112

    const int half = lane >> 4;         // 0/1
    const int lrow = lane & 15;

    v8f acc[4][2];
    #pragma unroll
    for (int r = 0; r < 4; r++)
        for (int c = 0; c < 2; c++) acc[r][c] = zero8();

    for (int k0 = 0; k0 < K; k0 += BK) {
        // A tile 128x32 -> LDS (vectorized 2x uint4 per thread)
        {
            const uint4* gp = reinterpret_cast<const uint4*>(A + (size_t)(m0 + a_row) * K + k0 + a_ks);
            uint4 x0 = gp[0], x1 = gp[1];
            uint4* sp = reinterpret_cast<uint4*>(&As[a_row * LSTR + a_ks]);
            sp[0] = x0; sp[1] = x1;
        }
        // B tile 32x128 -> LDS transposed Bs[n][k]
        {
            const uint4* gp = reinterpret_cast<const uint4*>(B + (size_t)(k0 + b_krow) * N + n0 + b_ns);
            U4u16 y0, y1; y0.q = gp[0]; y1.q = gp[1];
            #pragma unroll
            for (int j = 0; j < 8; j++) Bs[(b_ns + j)     * LSTR + b_krow] = y0.s[j];
            #pragma unroll
            for (int j = 0; j < 8; j++) Bs[(b_ns + 8 + j) * LSTR + b_krow] = y1.s[j];
        }
        __syncthreads();

        // B fragments (32x16, K pairs contiguous in Bs[n][k])
        Frag bf[2];
        #pragma unroll
        for (int c = 0; c < 2; c++) {
            int col = wc * 32 + c * 16 + lrow;
            int kb  = half * 16;
            #pragma unroll
            for (int v = 0; v < 8; v++)
                bf[c].u[v] = *reinterpret_cast<const unsigned int*>(&Bs[col * LSTR + kb + 2 * v]);
        }
        #pragma unroll
        for (int r = 0; r < 4; r++) {
            Frag af;
            int row = wr * 64 + r * 16 + lrow;
            int kb  = half * 16;
            #pragma unroll
            for (int v = 0; v < 8; v++) {
                int k = kb + 2 * v;
                af.u[v] = *reinterpret_cast<const unsigned int*>(&As[row * LSTR + k]);
            }
            #pragma unroll
            for (int c = 0; c < 2; c++)
                acc[r][c] = __builtin_amdgcn_wmma_f32_16x16x32_bf16(
                    false, af.v, false, bf[c].v, (short)0, acc[r][c], false, false);
        }
        __syncthreads();
    }

    // epilogue: D layout row = v + 8*half, col = lrow
    #pragma unroll
    for (int r = 0; r < 4; r++) {
        int grow = m0 + wr * 64 + r * 16 + half * 8;
        #pragma unroll
        for (int c = 0; c < 2; c++) {
            int gcol = n0 + wc * 32 + c * 16 + lrow;
            float bv = bias[gcol];
            #pragma unroll
            for (int v = 0; v < 8; v++)
                store_out(C, (size_t)(grow + v) * N + gcol, acc[r][c][v] + bv);
        }
    }
}

// ---------------- RoPE + repack ----------------
// qkv bf16 [4096, 7680] -> Qp/Kp bf16 [B,H,S,96] (pad zeroed), Vt bf16 [B,H,80,S]
__global__ __launch_bounds__(256)
void rope_pack(const unsigned short* __restrict__ qkv,
               const int* __restrict__ pos_ids,
               unsigned short* __restrict__ Qp,
               unsigned short* __restrict__ Kp,
               unsigned short* __restrict__ Vt) {
    size_t tid = (size_t)blockIdx.x * blockDim.x + threadIdx.x;
    const size_t total = (size_t)BB * NHEADS * SS * HDP;
    if (tid >= total) return;
    int d = (int)(tid % HDP);
    int s = (int)((tid / HDP) % SS);
    int h = (int)((tid / ((size_t)HDP * SS)) % NHEADS);
    int b = (int)(tid / ((size_t)HDP * SS * NHEADS));

    size_t qidx = ((size_t)(b * NHEADS + h) * SS + s) * HDP + d;
    if (d >= HDIM) { Qp[qidx] = 0; Kp[qidx] = 0; return; }

    size_t base = ((size_t)(b * SS + s)) * QKV_N + h * HDIM + d;
    float q = bf2f(qkv[base]);
    float k = bf2f(qkv[base + HID]);
    float v = bf2f(qkv[base + 2 * HID]);

    if (d < 32) {
        float pos = (float)pos_ids[b * SS + s];
        int   j   = (d < 16) ? d : d - 16;
        float inv = powf(10000.f, -(float)j / 16.f);
        float ang = pos * inv;
        float cs = cosf(ang), sn = sinf(ang);
        if (d < 16) {
            float q2 = bf2f(qkv[base + 16]);
            float k2 = bf2f(qkv[base + HID + 16]);
            q = q * cs - q2 * sn;
            k = k * cs - k2 * sn;
        } else {
            float q1 = bf2f(qkv[base - 16]);
            float k1 = bf2f(qkv[base + HID - 16]);
            q = q * cs + q1 * sn;
            k = k * cs + k1 * sn;
        }
    }
    Qp[qidx] = f2bf(q);
    Kp[qidx] = f2bf(k);
    Vt[((size_t)(b * NHEADS + h) * HDIM + d) * SS + s] = f2bf(v);
}

// ---------------- causal flash attention ----------------
// grid (B*H, S/64), 128 threads = 4 waves; each wave owns a 16-row Q strip.
__global__ __launch_bounds__(128)
void flash_attn(const unsigned short* __restrict__ Qp,
                const unsigned short* __restrict__ Kp,
                const unsigned short* __restrict__ Vt,
                unsigned short* __restrict__ attn) {
    __shared__ unsigned short Ps[4][16 * 32];   // wave-private P staging (D->A layout)

    const int t    = threadIdx.x;
    const int lane = t & 31;
    const int w    = t >> 5;
    const int bh   = blockIdx.x;
    const int b    = bh / NHEADS;
    const int h    = bh % NHEADS;
    const int q0   = blockIdx.y * 64 + w * 16;
    const int half = lane >> 4;
    const int lrow = lane & 15;

    const size_t qk_base = (size_t)bh * SS * HDP;
    const size_t v_base  = (size_t)bh * HDIM * SS;

    // Q A-fragments: 3 K-chunks of 32 over head dim (80 padded to 96)
    Frag qf[3];
    {
        int row = q0 + lrow;
        int kb  = half * 16;
        #pragma unroll
        for (int ch = 0; ch < 3; ch++)
            #pragma unroll
            for (int v = 0; v < 8; v++) {
                int k = ch * 32 + kb + 2 * v;
                qf[ch].u[v] = *reinterpret_cast<const unsigned int*>(
                    &Qp[qk_base + (size_t)row * HDP + k]);
            }
    }

    v8f acco[5];
    #pragma unroll
    for (int tt = 0; tt < 5; tt++) acco[tt] = zero8();
    float mrow[8], lsum[8];
    #pragma unroll
    for (int v = 0; v < 8; v++) { mrow[v] = -3.0e38f; lsum[v] = 0.f; }

    const float scale = 0.11180339887498949f;   // 80^-0.5
    const int nkb = ((q0 + 15) >> 5) + 1;       // causal: keys <= q0+15

    for (int kb = 0; kb < nkb; kb++) {
        const int kstart = kb * 32;

        // scores: 16x32 = 2 N-tiles x 3 K-chunks of WMMA
        v8f sacc[2]; sacc[0] = zero8(); sacc[1] = zero8();
        #pragma unroll
        for (int c = 0; c < 2; c++) {
            int key = kstart + c * 16 + lrow;
            int kkb = half * 16;
            #pragma unroll
            for (int ch = 0; ch < 3; ch++) {
                Frag kf;
                #pragma unroll
                for (int v = 0; v < 8; v++) {
                    int k = ch * 32 + kkb + 2 * v;
                    kf.u[v] = *reinterpret_cast<const unsigned int*>(
                        &Kp[qk_base + (size_t)key * HDP + k]);
                }
                sacc[c] = __builtin_amdgcn_wmma_f32_16x16x32_bf16(
                    false, qf[ch].v, false, kf.v, (short)0, sacc[c], false, false);
            }
        }

        // scale + causal mask + online softmax (rows live across 16 lanes of a half)
        #pragma unroll
        for (int v = 0; v < 8; v++) {
            int qrow = q0 + v + 8 * half;
            float p0, p1;
            {
                int key0 = kstart + lrow, key1 = kstart + 16 + lrow;
                p0 = sacc[0][v] * scale; if (key0 > qrow) p0 = -3.0e38f;
                p1 = sacc[1][v] * scale; if (key1 > qrow) p1 = -3.0e38f;
            }
            float rmax = fmaxf(p0, p1);
            #pragma unroll
            for (int m = 8; m >= 1; m >>= 1) rmax = fmaxf(rmax, __shfl_xor(rmax, m, 16));
            float mnew = fmaxf(mrow[v], rmax);
            float corr = __expf(mrow[v] - mnew);
            mrow[v] = mnew;
            p0 = __expf(p0 - mnew);
            p1 = __expf(p1 - mnew);
            float rsum = p0 + p1;
            #pragma unroll
            for (int m = 8; m >= 1; m >>= 1) rsum += __shfl_xor(rsum, m, 16);
            lsum[v] = lsum[v] * corr + rsum;
            #pragma unroll
            for (int tt = 0; tt < 5; tt++) acco[tt][v] *= corr;
            // stage P (D layout -> LDS row-major 16x32)
            int prow = v + 8 * half;
            Ps[w][prow * 32 + lrow]      = f2bf(p0);
            Ps[w][prow * 32 + 16 + lrow] = f2bf(p1);
        }

        // read P back as a 16x32 A-fragment (same-wave LDS ops are in order)
        Frag pf;
        {
            int kba = half * 16;
            #pragma unroll
            for (int v = 0; v < 8; v++) {
                int k = kba + 2 * v;
                pf.u[v] = *reinterpret_cast<const unsigned int*>(&Ps[w][lrow * 32 + k]);
            }
        }

        // O += P * V : 5 head-dim tiles, V^T gives contiguous b32 key-pair loads
        int kkb = half * 16;
        #pragma unroll
        for (int tt = 0; tt < 5; tt++) {
            Frag vf;
            int hd = tt * 16 + lrow;
            #pragma unroll
            for (int v = 0; v < 8; v++) {
                int key = kstart + kkb + 2 * v;
                vf.u[v] = *reinterpret_cast<const unsigned int*>(
                    &Vt[v_base + (size_t)hd * SS + key]);
            }
            acco[tt] = __builtin_amdgcn_wmma_f32_16x16x32_bf16(
                false, pf.v, false, vf.v, (short)0, acco[tt], false, false);
        }
    }

    // normalize + store attn bf16 [B,S,2560]
    #pragma unroll
    for (int tt = 0; tt < 5; tt++)
        #pragma unroll
        for (int v = 0; v < 8; v++) {
            int qrow = q0 + v + 8 * half;
            float o = acco[tt][v] / lsum[v];
            attn[((size_t)(b * SS + qrow)) * HID + h * HDIM + tt * 16 + lrow] = f2bf(o);
        }
}

// ---------------- launcher ----------------
extern "C" void kernel_launch(void* const* d_in, const int* in_sizes, int n_in,
                              void* d_out, int out_size, void* d_ws, size_t ws_size,
                              hipStream_t stream) {
    const int*   pos_ids = (const int*)d_in[0];
    const float* hidden  = (const float*)d_in[1];
    const float* w_qkv   = (const float*)d_in[2];
    const float* b_qkv   = (const float*)d_in[3];
    const float* w_dense = (const float*)d_in[4];
    const float* b_dense = (const float*)d_in[5];
    float* out = (float*)d_out;

    unsigned short* ws16 = (unsigned short*)d_ws;
    size_t off = 0;
    auto alloc16 = [&](size_t n) { unsigned short* p = ws16 + off; off += n; return p; };

    unsigned short* wq_b   = alloc16((size_t)HID * QKV_N);     // 19,660,800
    unsigned short* wd_b   = alloc16((size_t)HID * HID);       //  6,553,600
    unsigned short* hid_b  = alloc16((size_t)MROWS * HID);     // 10,485,760
    unsigned short* qkv_b  = alloc16((size_t)MROWS * QKV_N);   // 31,457,280
    unsigned short* Qp     = alloc16((size_t)BB * NHEADS * SS * HDP);
    unsigned short* Kp     = alloc16((size_t)BB * NHEADS * SS * HDP);
    unsigned short* Vt     = alloc16((size_t)BB * NHEADS * HDIM * SS);
    unsigned short* attn_b = alloc16((size_t)MROWS * HID);

    // 1) convert fp32 -> bf16
    {
        size_t n1 = (size_t)HID * QKV_N;
        cvt_f32_bf16<<<(unsigned)((n1 + 255) / 256), 256, 0, stream>>>(w_qkv, wq_b, n1);
        size_t n2 = (size_t)HID * HID;
        cvt_f32_bf16<<<(unsigned)((n2 + 255) / 256), 256, 0, stream>>>(w_dense, wd_b, n2);
        size_t n3 = (size_t)MROWS * HID;
        cvt_f32_bf16<<<(unsigned)((n3 + 255) / 256), 256, 0, stream>>>(hidden, hid_b, n3);
    }
    // 2) QKV GEMM (bf16 out)
    gemm_bf16<unsigned short><<<dim3(QKV_N / BN, MROWS / BM), 256, 0, stream>>>(
        hid_b, wq_b, b_qkv, qkv_b, MROWS, QKV_N, HID);
    // 3) RoPE + repack
    {
        size_t n = (size_t)BB * NHEADS * SS * HDP;
        rope_pack<<<(unsigned)((n + 255) / 256), 256, 0, stream>>>(qkv_b, pos_ids, Qp, Kp, Vt);
    }
    // 4) causal flash attention
    flash_attn<<<dim3(BB * NHEADS, SS / 64), 128, 0, stream>>>(Qp, Kp, Vt, attn_b);
    // 5) dense projection (fp32 out)
    gemm_bf16<float><<<dim3(HID / BN, MROWS / BM), 256, 0, stream>>>(
        attn_b, wd_b, b_dense, out, MROWS, HID, HID);
}
